// GroupedMoEExperts_72636486910163
// MI455X (gfx1250) — compile-verified, matrix operand order checked by reading, not run on hardware
//
#include <hip/hip_runtime.h>

// ---------------- problem constants (from reference) ----------------
constexpr int T_ = 4096;
constexpr int K_ = 8;
constexpr int E_ = 64;
constexpr int D_ = 2048;
constexpr int F_ = 512;
constexpr int C_ = 768;
constexpr int N_ = T_ * K_;   // 32768 routed rows

// ---------------- CDNA5 WMMA types ----------------
typedef __attribute__((ext_vector_type(16))) __bf16        v16bf;
typedef __attribute__((ext_vector_type(8)))  float         v8f;
typedef __attribute__((ext_vector_type(4)))  unsigned int  v4u;

union Frag { v4u u[2]; v16bf b; };

__device__ __forceinline__ unsigned short f2bf(float f) {
    unsigned int u = __builtin_bit_cast(unsigned int, f);
    u += 0x7FFFu + ((u >> 16) & 1u);   // round-to-nearest-even
    return (unsigned short)(u >> 16);
}

// async global->LDS copy (CDNA5), tracked by ASYNCcnt
__device__ __forceinline__ void async_copy_b128(unsigned lds_off, const void* gaddr) {
    asm volatile("global_load_async_to_lds_b128 %0, %1, off"
                 :: "v"(lds_off), "v"(gaddr)
                 : "memory");
}

// Split barriers that wait ONLY on the counters correctness needs.
// (__syncthreads() lowers to a fence that drains LOADcnt too, which would
// serialize our software-pipelined global loads against the barrier.)
__device__ __forceinline__ void barrier_ds() {
    asm volatile("s_wait_dscnt 0x0\n\t"
                 "s_barrier_signal -1\n\t"
                 "s_barrier_wait -1" ::: "memory");
}
__device__ __forceinline__ void barrier_ds_async() {
    asm volatile("s_wait_asynccnt 0x0\n\t"
                 "s_wait_dscnt 0x0\n\t"
                 "s_barrier_signal -1\n\t"
                 "s_barrier_wait -1" ::: "memory");
}

// ---------------- staging helpers ----------------
__device__ __forceinline__ void stage_a_regs(float4 ar[4], const float* xrow, int kb, int ahalf) {
    if (xrow) {
        const float4* p = (const float4*)(xrow + kb + ahalf * 16);
#pragma unroll
        for (int i = 0; i < 4; ++i) ar[i] = p[i];
    } else {
#pragma unroll
        for (int i = 0; i < 4; ++i) ar[i] = make_float4(0.f, 0.f, 0.f, 0.f);
    }
}
__device__ __forceinline__ void store_a_tile(unsigned short (*At)[40], int arow, int ahalf,
                                             const float4 ar[4]) {
    unsigned int* dst = (unsigned int*)&At[arow][ahalf * 16];
#pragma unroll
    for (int i = 0; i < 4; ++i) {
        dst[2 * i + 0] = (unsigned)f2bf(ar[i].x) | ((unsigned)f2bf(ar[i].y) << 16);
        dst[2 * i + 1] = (unsigned)f2bf(ar[i].z) | ((unsigned)f2bf(ar[i].w) << 16);
    }
}
__device__ __forceinline__ void stage_b_regs(float4 br[4], const float* wbase, int kb, int ldk) {
    const float4* p = (const float4*)(wbase + (size_t)kb * ldk);
#pragma unroll
    for (int i = 0; i < 4; ++i) br[i] = p[i];
}
__device__ __forceinline__ void store_b_tile(unsigned short (*Bt)[40], int bc, int bk,
                                             const float4 br[4]) {
#pragma unroll
    for (int i = 0; i < 4; ++i) {
        Bt[bc + 4 * i + 0][bk] = f2bf(br[i].x);
        Bt[bc + 4 * i + 1][bk] = f2bf(br[i].y);
        Bt[bc + 4 * i + 2][bk] = f2bf(br[i].z);
        Bt[bc + 4 * i + 3][bk] = f2bf(br[i].w);
    }
}

// wave computes 16(M) x 128(N) with 8 WMMA accumulators; B fragments double-buffered
__device__ __forceinline__ void compute_tile(const unsigned short (*At)[40],
                                             const unsigned short (*Bt)[40],
                                             int wave, int li, int kh, v8f acc[8]) {
    Frag a;
    a.u[0] = *(const v4u*)&At[wave * 16 + li][kh * 8];
    a.u[1] = *(const v4u*)&At[wave * 16 + li][16 + kh * 8];
    Frag b[2];
    b[0].u[0] = *(const v4u*)&Bt[li][kh * 8];
    b[0].u[1] = *(const v4u*)&Bt[li][16 + kh * 8];
#pragma unroll
    for (int j = 0; j < 8; ++j) {
        if (j < 7) {
            b[(j + 1) & 1].u[0] = *(const v4u*)&Bt[(j + 1) * 16 + li][kh * 8];
            b[(j + 1) & 1].u[1] = *(const v4u*)&Bt[(j + 1) * 16 + li][16 + kh * 8];
        }
        acc[j] = __builtin_amdgcn_wmma_f32_16x16x32_bf16(
            false, a.b, false, b[j & 1].b, (short)0, acc[j], false, false);
    }
}

// ---------------- kernel 0: zero output ----------------
__global__ void moe_zero_f32(float* __restrict__ p, int n) {
    int i = blockIdx.x * 256 + threadIdx.x;
    if (i < n) p[i] = 0.0f;
}

// ---------------- kernel 1: routing (stable, one wave per expert) ----------------
__global__ __launch_bounds__(32) void moe_route(
    const int*   __restrict__ indices,
    const float* __restrict__ weights,
    int*         __restrict__ row_tok,
    float*       __restrict__ row_w)
{
    const int e    = blockIdx.x;
    const int lane = threadIdx.x;

    for (int i = lane; i < C_; i += 32) {
        row_tok[e * C_ + i] = -1;
        row_w  [e * C_ + i] = 0.0f;
    }
    int base = 0;
    for (int n0 = 0; n0 < N_; n0 += 32) {
        const int n = n0 + lane;
        const bool match = (indices[n] == e);
        const unsigned mask  = __builtin_amdgcn_ballot_w32(match);
        const unsigned below = mask & ((1u << lane) - 1u);
        const int pos = base + __popc(below);
        if (match && pos < C_) {
            row_tok[e * C_ + pos] = n >> 3;   // token = n / K (K == 8)
            row_w  [e * C_ + pos] = weights[n];
        }
        base += __popc(mask);
    }
}

// ---------------- kernel 2: up GEMM + relu^2, h stored bf16 ----------------
// grid: (F/128, C/128, E), block: 256 (8 waves). Double-buffered LDS pipeline.
__global__ __launch_bounds__(256) void moe_up(
    const float*  __restrict__ x,        // [T, D]
    const float*  __restrict__ up_w,     // [E, D, F]
    const int*    __restrict__ row_tok,  // [E, C]
    unsigned short* __restrict__ h)      // [E, C, F] bf16
{
    __shared__ __align__(16) unsigned short At[2][128][40];
    __shared__ __align__(16) unsigned short Bt[2][128][40];

    const int e     = blockIdx.z;
    const int crow0 = blockIdx.y * 128;
    const int nb    = blockIdx.x * 128;
    const int tid   = threadIdx.x;
    const int wave  = tid >> 5;
    const int lane  = tid & 31;
    const int li    = lane & 15;
    const int kh    = lane >> 4;

    const int arow  = tid >> 1;
    const int ahalf = tid & 1;
    const int tok   = row_tok[e * C_ + crow0 + arow];
    const float* xrow = (tok >= 0) ? (x + (size_t)tok * D_) : nullptr;

    const int bk = tid >> 3;
    const int bc = (tid & 7) * 16;
    const float* wbase = up_w + (size_t)e * D_ * F_ + (size_t)bk * F_ + nb + bc;

    v8f acc[8];
#pragma unroll
    for (int j = 0; j < 8; ++j) acc[j] = (v8f)0.0f;

    float4 ar[4], br[4];
    stage_a_regs(ar, xrow, 0, ahalf);
    stage_b_regs(br, wbase, 0, F_);

    int cur = 0;
    for (int kb = 0; kb < D_; kb += 32, cur ^= 1) {
        store_a_tile(At[cur], arow, ahalf, ar);
        store_b_tile(Bt[cur], bc, bk, br);
        const int kbn = (kb + 32) & (D_ - 1);           // wraps harmlessly on last iter
        stage_a_regs(ar, xrow, kbn, ahalf);
        stage_b_regs(br, wbase, kbn, F_);
        __builtin_prefetch(wbase + (size_t)((kb + 64) & (D_ - 1)) * F_, 0, 0);
        barrier_ds();                                    // waits DScnt only
        compute_tile(At[cur], Bt[cur], wave, li, kh, acc);
    }

    // epilogue: relu^2, store h bf16
#pragma unroll
    for (int j = 0; j < 8; ++j) {
#pragma unroll
        for (int v = 0; v < 8; ++v) {
            const int m = kh * 8 + v;
            const int n = j * 16 + li;
            float val = acc[j][v];
            val = (val > 0.0f) ? val * val : 0.0f;
            h[(size_t)(e * C_ + crow0 + wave * 16 + m) * F_ + nb + n] = f2bf(val);
        }
    }
}

// ---------------- kernel 3: down GEMM + weighted atomic combine ----------------
// grid: (D/128, C/128, E), block: 256 (8 waves).
// A tile (bf16 h) staged via global_load_async_to_lds_b128 (ASYNCcnt),
// B tile (fp32->bf16) via register pipeline. Double-buffered LDS.
__global__ __launch_bounds__(256) void moe_down(
    const unsigned short* __restrict__ h,       // [E, C, F] bf16
    const float*          __restrict__ down_w,  // [E, F, D]
    const int*            __restrict__ row_tok, // [E, C]
    const float*          __restrict__ row_w,   // [E, C]
    float*                __restrict__ y)       // [T, D]
{
    __shared__ __align__(16) unsigned short At[2][128][40];
    __shared__ __align__(16) unsigned short Bt[2][128][40];

    const int e     = blockIdx.z;
    const int crow0 = blockIdx.y * 128;
    const int nb    = blockIdx.x * 128;
    const int tid   = threadIdx.x;
    const int wave  = tid >> 5;
    const int lane  = tid & 31;
    const int li    = lane & 15;
    const int kh    = lane >> 4;

    const int arow  = tid >> 1;
    const int ahalf = tid & 1;
    const unsigned short* gsrc0 = h + (size_t)(e * C_ + crow0 + arow) * F_ + ahalf * 16;

    const int bk = tid >> 3;
    const int bc = (tid & 7) * 16;
    const float* wbase = down_w + (size_t)e * F_ * D_ + (size_t)bk * D_ + nb + bc;

    unsigned lds_a[2];
    lds_a[0] = (unsigned)(uintptr_t)&At[0][arow][ahalf * 16];
    lds_a[1] = (unsigned)(uintptr_t)&At[1][arow][ahalf * 16];

    v8f acc[8];
#pragma unroll
    for (int j = 0; j < 8; ++j) acc[j] = (v8f)0.0f;

    // prologue: async A(kb=0) -> buffer 0, B(kb=0) -> regs
    async_copy_b128(lds_a[0],      gsrc0);
    async_copy_b128(lds_a[0] + 16, gsrc0 + 8);
    float4 br[4];
    stage_b_regs(br, wbase, 0, D_);

    int cur = 0;
    for (int kb = 0; kb < F_; kb += 32, cur ^= 1) {
        store_b_tile(Bt[cur], bc, bk, br);
        const int kbn = (kb + 32) & (F_ - 1);           // wraps harmlessly on last iter
        stage_b_regs(br, wbase, kbn, D_);
        __builtin_prefetch(wbase + (size_t)((kb + 64) & (F_ - 1)) * D_, 0, 0);
        barrier_ds_async();                              // waits ASYNCcnt + DScnt only
        // issue A(kb+32) into the other buffer; overlaps WMMA below.
        // Safe: all readers of LDS[cur^1] (iteration i-1) finished before this barrier.
        async_copy_b128(lds_a[cur ^ 1],      gsrc0 + kbn);
        async_copy_b128(lds_a[cur ^ 1] + 16, gsrc0 + kbn + 8);
        compute_tile(At[cur], Bt[cur], wave, li, kh, acc);
    }

    // epilogue: y[tok, :] += w * row
    const int slotbase = e * C_ + crow0 + wave * 16 + kh * 8;
#pragma unroll
    for (int v = 0; v < 8; ++v) {
        const int tok = row_tok[slotbase + v];
        if (tok < 0) continue;
        const float w = row_w[slotbase + v];
        float* yrow = y + (size_t)tok * D_ + nb;
#pragma unroll
        for (int j = 0; j < 8; ++j) {
            atomicAdd(&yrow[j * 16 + li], w * acc[j][v]);
        }
    }
}

// ---------------- host launcher ----------------
extern "C" void kernel_launch(void* const* d_in, const int* in_sizes, int n_in,
                              void* d_out, int out_size, void* d_ws, size_t ws_size,
                              hipStream_t stream) {
    const float* x       = (const float*)d_in[0];   // [T, D]
    const float* weights = (const float*)d_in[1];   // [T, K]
    const int*   indices = (const int*)  d_in[2];   // [T, K]
    const float* up_w    = (const float*)d_in[3];   // [E, D, F]
    const float* down_w  = (const float*)d_in[4];   // [E, F, D]
    float* y = (float*)d_out;                        // [T, D]

    char* ws = (char*)d_ws;
    int*            row_tok = (int*)ws;                           // E*C ints   (192 KB)
    float*          row_w   = (float*)(ws + (size_t)E_ * C_ * 4); // E*C floats (192 KB)
    unsigned short* hbuf    = (unsigned short*)(ws + (size_t)2 * E_ * C_ * 4); // E*C*F bf16 (~48 MB)

    moe_zero_f32<<<(T_ * D_ + 255) / 256, 256, 0, stream>>>(y, T_ * D_);
    moe_route<<<E_, 32, 0, stream>>>(indices, weights, row_tok, row_w);
    moe_up<<<dim3(F_ / 128, C_ / 128, E_), 256, 0, stream>>>(x, up_w, row_tok, hbuf);
    moe_down<<<dim3(D_ / 128, C_ / 128, E_), 256, 0, stream>>>(hbuf, down_w, row_tok, row_w, y);
}
